// SyntaxBasedDecoder_44040594653390
// MI455X (gfx1250) — matrix operand order repaired and checked by reading, not compiled, not run on hardware
//
#include <hip/hip_runtime.h>
#include <cmath>

// ---------------- problem dims ----------------
#define BB   32      // batch
#define TT   128     // decode steps
#define SS   256     // encoder length
#define HH   512     // hidden
#define AA   512     // attention dim
#define VV   512     // encoder value dim
#define RR   256     // rule vocab
#define VTT  32000   // token vocab
#define AEE  128     // action emb
#define NEE  32      // node emb
#define K0   (AEE + NEE + VV + HH)   // 1184  LSTM0 GEMM K
#define K1   (2 * HH)                // 1024  LSTM1 GEMM K
#define KO   HH                      // 512   logits GEMM K
#define NG   (4 * HH)                // 2048  gate width
#define NO   (RR + VTT)              // 32256 logits width

typedef __attribute__((ext_vector_type(16))) __bf16 v16bf;
typedef __attribute__((ext_vector_type(8)))  __bf16 v8bf;
typedef __attribute__((ext_vector_type(8)))  float  v8f;
typedef __attribute__((ext_vector_type(4)))  unsigned int v4u;
typedef __attribute__((ext_vector_type(8)))  int v8i;
typedef __attribute__((ext_vector_type(4)))  int v4i;

#if defined(__has_builtin)
#  if __has_builtin(__builtin_amdgcn_tensor_load_to_lds) && __has_builtin(__builtin_amdgcn_s_wait_tensorcnt)
#    define HAVE_TDM 1
#  endif
#endif

static __device__ __forceinline__ __bf16 to_bf(float f) {
  unsigned u = __builtin_bit_cast(unsigned, f);
  unsigned r = (u + 0x7FFFu + ((u >> 16) & 1u)) >> 16;
  return __builtin_bit_cast(__bf16, (unsigned short)r);
}
static __device__ __forceinline__ float sigm(float x) {
  return 1.0f / (1.0f + __expf(-x));
}
static __device__ __forceinline__ v16bf cat8(v8bf lo, v8bf hi) {
  v16bf o;
#pragma unroll
  for (int i = 0; i < 8; ++i) { o[i] = lo[i]; o[i + 8] = hi[i]; }
  return o;
}

// ---------------------------------------------------------------------------
// Pack fp32 weights (two matrices concatenated along K) into bf16
// WMMA-B fragment-major layout:
//   element (k,n) -> Bp[ (n/16)*(K/32)*512 + (k/32)*512 + (k%32/16)*256 + (n%16)*16 + (k%16) ]
// so a wave's 32x16 B-fragment for one k-chunk is 1KB contiguous, lane-coalesced.
// ---------------------------------------------------------------------------
__global__ void pack_catK(const float* __restrict__ W1, const float* __restrict__ W2,
                          int K1_, int K, int N, __bf16* __restrict__ Bp) {
  size_t idx = (size_t)blockIdx.x * 256 + threadIdx.x;
  if (idx >= (size_t)K * N) return;
  int k = (int)(idx / N), n = (int)(idx % N);
  float v = (k < K1_) ? W1[(size_t)k * N + n] : W2[(size_t)(k - K1_) * N + n];
  int kc = k >> 5, kr = k & 31, half = kr >> 4, j = kr & 15;
  int nt = n >> 4, r = n & 15;
  Bp[((size_t)nt * (K >> 5) + kc) * 512 + half * 256 + r * 16 + j] = to_bf(v);
}

// same, but concatenation along N:  [W1 (K x N1) | W2 (K x (N-N1))]
__global__ void pack_catN(const float* __restrict__ W1, const float* __restrict__ W2,
                          int N1, int N, int K, __bf16* __restrict__ Bp) {
  size_t idx = (size_t)blockIdx.x * 256 + threadIdx.x;
  if (idx >= (size_t)K * N) return;
  int k = (int)(idx / N), n = (int)(idx % N);
  float v = (n < N1) ? W1[(size_t)k * N1 + n] : W2[(size_t)k * (N - N1) + (n - N1)];
  int kc = k >> 5, kr = k & 31, half = kr >> 4, j = kr & 15;
  int nt = n >> 4, r = n & 15;
  Bp[((size_t)nt * (K >> 5) + kc) * 512 + half * 256 + r * 16 + j] = to_bf(v);
}

__global__ void bias_cat(const float* __restrict__ b1, const float* __restrict__ b2,
                         int N1, int N, float* __restrict__ out) {
  int n = blockIdx.x * 256 + threadIdx.x;
  if (n < N) out[n] = (n < N1) ? b1[n] : b2[n - N1];
}

// ---------------------------------------------------------------------------
// bf16 WMMA GEMM:  C[32,N] = A[32,K] * B[K,N] (+bias), fp32 accumulate.
// A: bf16 row-major, staged to LDS by the Tensor Data Mover (one 1-row tile,
//    data_size=8B, tile_dim0 = 8*K units), fragments read with ds_load_b128.
// B: packed fragment-major bf16 (contiguous coalesced v16bf loads) with
//    global_prefetch_b8 run-ahead.
// Block: 256 threads = 8 waves, each wave owns 32 N-columns (2 N-tiles),
// both 16-row M-tiles -> 4 v_wmma_f32_16x16x32_bf16 per 32-deep K chunk.
// ---------------------------------------------------------------------------
__global__ void gemm_bf16_wmma(const __bf16* __restrict__ A,
                               const __bf16* __restrict__ Bp,
                               const float* __restrict__ bias,
                               float* __restrict__ C, int K, int N) {
  extern __shared__ char smem[];
  __bf16* sA = (__bf16*)smem;                    // [32 * K] bf16
  const int tid = threadIdx.x;

#ifdef HAVE_TDM
  // ---- Tensor Data Mover: DMA the whole A panel (32*K*2 bytes) into LDS ----
  if (tid < 32) {   // wave 0 issues the descriptor; TDM ignores EXEC
    const unsigned long long ga = (unsigned long long)A;
    const unsigned Nu = (unsigned)(8 * K);       // # of 8-byte units (fits tile_dim0)
    v4u g0;
    g0[0] = 1u;                                  // count=1, no gather
    g0[1] = 0u;                                  // lds_addr = dynamic-LDS base
    g0[2] = (unsigned)(ga & 0xffffffffu);        // global_addr[31:0]
    g0[3] = (unsigned)((ga >> 32) & 0x01ffffffu) | 0x80000000u;  // addr[56:32] | type=2
    v8i g1;
    g1[0] = (int)0x00030000u;                    // data_size=3 (8B), no multicast/pad
    g1[1] = (int)((Nu & 0xffffu) << 16);         // tensor_dim0[15:0]
    g1[2] = (int)(((Nu >> 16) & 0xffffu) | (1u << 16)); // tensor_dim0[31:16], tensor_dim1=1
    g1[3] = (int)((Nu & 0xffffu) << 16);         // tile_dim0 = Nu
    g1[4] = 1;                                   // tile_dim1 = 1, tile_dim2 = 0
    g1[5] = (int)Nu;                             // tensor_dim0_stride[31:0]
    g1[6] = 0;                                   // stride hi / dim1_stride lo
    g1[7] = 0;
    v4i z4 = {0, 0, 0, 0};
#if __clang_major__ >= 23
    v8i z8 = {0, 0, 0, 0, 0, 0, 0, 0};
    __builtin_amdgcn_tensor_load_to_lds(g0, g1, z4, z4, z8, 0);
#else
    __builtin_amdgcn_tensor_load_to_lds(g0, g1, z4, z4, 0);
#endif
    __builtin_amdgcn_s_wait_tensorcnt(0);
  }
#else
  // fallback: cooperative stage of A into LDS
  {
    const unsigned* Au = (const unsigned*)A;
    unsigned* sAu = (unsigned*)sA;
    const int nU32 = (32 * K) >> 1;
    for (int i = tid; i < nU32; i += 256) sAu[i] = Au[i];
  }
#endif
  __syncthreads();

  const int wave = tid >> 5;
  const int lane = tid & 31;
  const int half = lane >> 4;
  const int r    = lane & 15;
  const int n0   = blockIdx.x * 256 + wave * 32;
  if (n0 >= N) return;
  const int Kc = K >> 5;

  v8f acc00 = {}, acc01 = {}, acc10 = {}, acc11 = {};
  const __bf16* bp0 = Bp + ((size_t)(n0 >> 4) * Kc) * 512 + lane * 16;
  const __bf16* bp1 = bp0 + (size_t)Kc * 512;
  const __bf16* a0p = sA + (size_t)r * K + half * 16;          // M-tile 0, row r
  const __bf16* a1p = sA + (size_t)(16 + r) * K + half * 16;   // M-tile 1, row 16+r

#pragma unroll 2
  for (int kc = 0; kc < Kc; ++kc) {
    const int k0 = kc << 5;
    if (kc + 4 < Kc) {   // run-ahead prefetch of the packed-B stream (L2-resident)
      __builtin_prefetch(bp0 + (size_t)(kc + 4) * 512, 0, 0);
      __builtin_prefetch(bp1 + (size_t)(kc + 4) * 512, 0, 0);
    }
    // A fragments (contiguous k order matching packed B: lanes 0-15 K{0..15}, 16-31 K{16..31})
    v16bf a0 = cat8(*(const v8bf*)(a0p + k0), *(const v8bf*)(a0p + k0 + 8));
    v16bf a1 = cat8(*(const v8bf*)(a1p + k0), *(const v8bf*)(a1p + k0 + 8));
    // B fragments: contiguous packed 1KB per wave-tile
    v16bf b0 = *(const v16bf*)(bp0 + (size_t)kc * 512);
    v16bf b1 = *(const v16bf*)(bp1 + (size_t)kc * 512);
    acc00 = __builtin_amdgcn_wmma_f32_16x16x32_bf16(false, a0, false, b0, (short)0, acc00, false, false);
    acc01 = __builtin_amdgcn_wmma_f32_16x16x32_bf16(false, a0, false, b1, (short)0, acc01, false, false);
    acc10 = __builtin_amdgcn_wmma_f32_16x16x32_bf16(false, a1, false, b0, (short)0, acc10, false, false);
    acc11 = __builtin_amdgcn_wmma_f32_16x16x32_bf16(false, a1, false, b1, (short)0, acc11, false, false);
  }

  // D layout: VGPR v -> row (v + half*8), col n0 + r  (per 16x16 f32 C/D table)
  const int c0 = n0 + r, c1 = n0 + 16 + r;
  const float bi0 = bias ? bias[c0] : 0.0f;
  const float bi1 = bias ? bias[c1] : 0.0f;
#pragma unroll
  for (int v = 0; v < 8; ++v) {
    const int row0 = half * 8 + v;
    const int row1 = 16 + row0;
    C[(size_t)row0 * N + c0] = acc00[v] + bi0;
    C[(size_t)row0 * N + c1] = acc01[v] + bi1;
    C[(size_t)row1 * N + c0] = acc10[v] + bi0;
    C[(size_t)row1 * N + c1] = acc11[v] + bi1;
  }
}

// ---------------------------------------------------------------------------
// Attention + embedding gather: one block per batch row (32 blocks, 256 thr).
// Writes bf16 LSTM0 input columns [0, 672) of xin0 (cols [672,1184) hold h0,
// maintained by the LSTM0 pointwise kernel).
// ---------------------------------------------------------------------------
__global__ void attn_kernel(const float* __restrict__ h1, const float* __restrict__ Wq,
                            const float* __restrict__ we, const float* __restrict__ pk,
                            const float* __restrict__ enc, const unsigned char* __restrict__ mask,
                            const float* __restrict__ rule_emb, const float* __restrict__ tok_emb,
                            const float* __restrict__ node_emb, const int* __restrict__ n2a,
                            const int* __restrict__ node_type, const int* __restrict__ action_value,
                            __bf16* __restrict__ xin0, int t) {
  __shared__ float pq[AA];
  __shared__ float sc[SS];
  __shared__ float red[SS];
  const int b = blockIdx.x, tid = threadIdx.x;
  const float* h = h1 + (size_t)b * HH;
  __bf16* xr = xin0 + (size_t)b * K0;

  // pq = h1[b] @ Wq   (coalesced column reads of Wq across threads)
  for (int a = tid; a < AA; a += 256) {
    float s = 0.0f;
    for (int k = 0; k < HH; ++k) s = fmaf(h[k], Wq[(size_t)k * AA + a], s);
    pq[a] = s;
  }
  __syncthreads();

  // scores[s] = sum_a tanh(pk + pq) * we
  {
    const int s = tid;
    const float* pks = pk + ((size_t)b * SS + s) * AA;
    float e = 0.0f;
    for (int a = 0; a < AA; ++a) e = fmaf(tanhf(pks[a] + pq[a]), we[a], e);
    if (!mask[(size_t)b * SS + s]) e = -INFINITY;
    sc[s] = e;
  }
  __syncthreads();

  // softmax over S=256
  red[tid] = sc[tid];
  __syncthreads();
  for (int off = 128; off > 0; off >>= 1) {
    if (tid < off) red[tid] = fmaxf(red[tid], red[tid + off]);
    __syncthreads();
  }
  const float mx = red[0];
  __syncthreads();
  const float ev = __expf(sc[tid] - mx);
  sc[tid] = ev; red[tid] = ev;
  __syncthreads();
  for (int off = 128; off > 0; off >>= 1) {
    if (tid < off) red[tid] += red[tid + off];
    __syncthreads();
  }
  const float inv = 1.0f / red[0];
  __syncthreads();
  sc[tid] *= inv;
  __syncthreads();

  // context[v] = sum_s p[s] * enc[b,s,v]  -> cols [160, 672)
  for (int v = tid; v < VV; v += 256) {
    float acc = 0.0f;
    for (int s = 0; s < SS; ++s) acc = fmaf(sc[s], enc[((size_t)b * SS + s) * VV + v], acc);
    xr[AEE + NEE + v] = to_bf(acc);
  }

  // previous-action embedding -> cols [0,128); node embedding -> cols [128,160)
  int pt = 0, pv = 0;
  if (t > 0) {
    const int ntp = node_type[(size_t)(t - 1) * BB + b];
    pt = n2a[ntp];
    pv = action_value[(size_t)(t - 1) * BB + b];
  }
  if (tid < AEE) {
    float pe = (pt == 0) ? rule_emb[(size_t)pv * AEE + tid]
             : (pt == 1) ? tok_emb[(size_t)pv * AEE + tid] : 0.0f;
    xr[tid] = to_bf(pe);
  } else if (tid < AEE + NEE) {
    const int nt = node_type[(size_t)t * BB + b];
    xr[tid] = to_bf(node_emb[(size_t)nt * NEE + (tid - AEE)]);
  }
}

// ---------------------------------------------------------------------------
// LSTM pointwise: gates z[32,2048] (i|f|g|o), fp32 cell state update,
// h written as bf16 directly into the next GEMM's input slots.
// ---------------------------------------------------------------------------
__global__ void lstm_cell(const float* __restrict__ z, float* __restrict__ c,
                          float* __restrict__ hf32,
                          __bf16* __restrict__ d0, int ld0,
                          __bf16* __restrict__ d1, int ld1) {
  const int b = blockIdx.x, j = threadIdx.x;     // 32 x 512
  const float* zb = z + (size_t)b * NG;
  const float i = sigm(zb[j]);
  const float f = sigm(zb[HH + j]);
  const float g = tanhf(zb[2 * HH + j]);
  const float o = sigm(zb[3 * HH + j]);
  const size_t ci = (size_t)b * HH + j;
  const float cn = f * c[ci] + i * g;
  c[ci] = cn;
  const float hv = o * tanhf(cn);
  if (hf32) hf32[ci] = hv;
  const __bf16 hb = to_bf(hv);
  d0[(size_t)b * ld0 + j] = hb;
  d1[(size_t)b * ld1 + j] = hb;
}

// ---------------------------------------------------------------------------
extern "C" void kernel_launch(void* const* d_in, const int* in_sizes, int n_in,
                              void* d_out, int out_size, void* d_ws, size_t ws_size,
                              hipStream_t stream) {
  const float* rule_emb = (const float*)d_in[0];
  const float* tok_emb  = (const float*)d_in[1];
  const float* node_emb = (const float*)d_in[2];
  const float* Wq  = (const float*)d_in[3];
  const float* we  = (const float*)d_in[4];
  const float* Wx0 = (const float*)d_in[5];
  const float* Wh0 = (const float*)d_in[6];
  const float* b0  = (const float*)d_in[7];
  const float* Wx1 = (const float*)d_in[8];
  const float* Wh1 = (const float*)d_in[9];
  const float* b1  = (const float*)d_in[10];
  const float* Wr  = (const float*)d_in[11];
  const float* br  = (const float*)d_in[12];
  const float* Wt  = (const float*)d_in[13];
  const float* bt  = (const float*)d_in[14];
  const float* enc = (const float*)d_in[15];
  const float* pk  = (const float*)d_in[16];
  const unsigned char* mask = (const unsigned char*)d_in[17];
  const int* n2a   = (const int*)d_in[18];
  const int* ntyp  = (const int*)d_in[19];
  const int* aval  = (const int*)d_in[20];
  float* out = (float*)d_out;

  // workspace carve-out (256B aligned)
  size_t off = 0;
  char* base = (char*)d_ws;
  auto alloc = [&](size_t bytes) -> char* {
    char* p = base + off;
    off += (bytes + 255) & ~(size_t)255;
    return p;
  };
  __bf16* Wp0  = (__bf16*)alloc((size_t)K0 * NG * 2);   // [Wx0;Wh0] packed
  __bf16* Wp1  = (__bf16*)alloc((size_t)K1 * NG * 2);   // [Wx1;Wh1] packed
  __bf16* Wpo  = (__bf16*)alloc((size_t)KO * NO * 2);   // [Wr|Wt] packed
  float*  bcat = (float*) alloc((size_t)NO * 4);
  float*  c0   = (float*) alloc((size_t)BB * HH * 4);
  float*  c1   = (float*) alloc((size_t)BB * HH * 4);
  float*  h1f  = (float*) alloc((size_t)BB * HH * 4);
  __bf16* xin0 = (__bf16*)alloc((size_t)BB * K0 * 2);   // [x | h0]
  __bf16* xin1 = (__bf16*)alloc((size_t)BB * K1 * 2);   // [h0 | h1]
  __bf16* h1bf = (__bf16*)alloc((size_t)BB * HH * 2);
  float*  z0   = (float*) alloc((size_t)BB * NG * 4);
  float*  z1   = (float*) alloc((size_t)BB * NG * 4);

  // one-time weight packing (deterministic; rerun every call)
  {
    size_t n;
    n = (size_t)K0 * NG;
    pack_catK<<<dim3((unsigned)((n + 255) / 256)), 256, 0, stream>>>(Wx0, Wh0, AEE + NEE + VV, K0, NG, Wp0);
    n = (size_t)K1 * NG;
    pack_catK<<<dim3((unsigned)((n + 255) / 256)), 256, 0, stream>>>(Wx1, Wh1, HH, K1, NG, Wp1);
    n = (size_t)KO * NO;
    pack_catN<<<dim3((unsigned)((n + 255) / 256)), 256, 0, stream>>>(Wr, Wt, RR, NO, KO, Wpo);
    bias_cat<<<dim3((NO + 255) / 256), 256, 0, stream>>>(br, bt, RR, NO, bcat);
  }

  // zero recurrent state (graph-capturable)
  hipMemsetAsync(c0,   0, (size_t)BB * HH * 4, stream);
  hipMemsetAsync(c1,   0, (size_t)BB * HH * 4, stream);
  hipMemsetAsync(h1f,  0, (size_t)BB * HH * 4, stream);
  hipMemsetAsync(xin0, 0, (size_t)BB * K0 * 2, stream);
  hipMemsetAsync(xin1, 0, (size_t)BB * K1 * 2, stream);
  hipMemsetAsync(h1bf, 0, (size_t)BB * HH * 2, stream);

  const size_t lds0 = (size_t)32 * K0 * 2;   // A staging LDS bytes
  const size_t lds1 = (size_t)32 * K1 * 2;
  const size_t ldso = (size_t)32 * KO * 2;

  for (int t = 0; t < TT; ++t) {
    attn_kernel<<<dim3(BB), 256, 0, stream>>>(h1f, Wq, we, pk, enc, mask,
                                              rule_emb, tok_emb, node_emb,
                                              n2a, ntyp, aval, xin0, t);
    // LSTM0 gates: z0 = [x|h0] @ Wp0 + b0
    gemm_bf16_wmma<<<dim3(NG / 256), 256, lds0, stream>>>(xin0, Wp0, b0, z0, K0, NG);
    lstm_cell<<<dim3(BB), HH, 0, stream>>>(z0, c0, nullptr,
                                           xin0 + (AEE + NEE + VV), K0,  // h0 -> xin0[:,672:]
                                           xin1, K1);                    // h0 -> xin1[:,0:512]
    // LSTM1 gates: z1 = [h0|h1] @ Wp1 + b1
    gemm_bf16_wmma<<<dim3(NG / 256), 256, lds1, stream>>>(xin1, Wp1, b1, z1, K1, NG);
    lstm_cell<<<dim3(BB), HH, 0, stream>>>(z1, c1, h1f,
                                           xin1 + HH, K1,               // h1 -> xin1[:,512:]
                                           h1bf, HH);                   // h1 -> logits input
    // logits: out[t] = h1 @ [Wr|Wt] + [br|bt]
    gemm_bf16_wmma<<<dim3(NO / 256), 256, ldso, stream>>>(h1bf, Wpo, bcat,
                                                          out + (size_t)t * BB * NO, KO, NO);
  }
}
